// Encoder_60455959659037
// MI455X (gfx1250) — matrix operand order repaired and checked by reading, not compiled, not run on hardware
//
#include <hip/hip_runtime.h>
#include <cstdint>
#include <cstddef>

// ---------------------------------------------------------------------------
// Types for WMMA fragments
// ---------------------------------------------------------------------------
typedef unsigned short u16t;
typedef __bf16 v16bf __attribute__((ext_vector_type(16)));
typedef float  v8f   __attribute__((ext_vector_type(8)));
typedef unsigned int v4u __attribute__((ext_vector_type(4)));
typedef unsigned int v8u __attribute__((ext_vector_type(8)));

union BFrag { v16bf bf; v8u u8; v4u u4[2]; };

static __device__ __forceinline__ u16t f2bf(float f) {
  unsigned u = __float_as_uint(f);
  unsigned r = (u + 0x7FFFu + ((u >> 16) & 1u)) >> 16;   // RNE
  return (u16t)r;
}
static __device__ __forceinline__ float sigmoidf_(float x) {
  return 1.0f / (1.0f + __expf(-x));
}
static __device__ __forceinline__ v8f wmma_bf16(const BFrag& a, const BFrag& b, v8f c) {
  return __builtin_amdgcn_wmma_f32_16x16x32_bf16(false, a.bf, false, b.bf,
                                                 (short)0, c, false, false);
}

#define NB   1024   // batch
#define NS   256    // sequence
#define HL   512    // LSTM hidden
#define BTL  16     // batch rows per workgroup
#define LDAA 1024   // A-panel stride (u16 elems) in LDS

// ---------------------------------------------------------------------------
// Weight packing: fp32 [K][N] -> bf16 [N][Kpad] (WMMA B-fragment friendly:
// each lane's 16 K-values are contiguous).  zrow inserts a zero K-row (used
// to pad layer-0 x dim 63 -> 64).
// ---------------------------------------------------------------------------
__global__ void k_pack_wT(const float* __restrict__ src, u16t* __restrict__ dst,
                          int K, int N, int Kpad, int zrow) {
  size_t i = (size_t)blockIdx.x * blockDim.x + threadIdx.x;
  size_t tot = (size_t)N * Kpad;
  if (i >= tot) return;
  int n = (int)(i / Kpad);
  int k = (int)(i % Kpad);
  float v = 0.f;
  if (k != zrow) {
    int ks = (zrow >= 0 && k > zrow) ? (k - 1) : k;
    if (ks < K) v = src[(size_t)ks * N + n];
  }
  dst[i] = f2bf(v);
}

// ---------------------------------------------------------------------------
// covl = (1-MCOVL)*AT + COVL*MCOVL, reversed over time, split into x (bf16,
// padded 63->64) and T = 1/log(dt + C2).
// ---------------------------------------------------------------------------
__global__ void k_prep_x(const float* __restrict__ COVL, const float* __restrict__ MCOVL,
                         const float* __restrict__ AT, u16t* __restrict__ xrev,
                         float* __restrict__ Trev) {
  size_t i = (size_t)blockIdx.x * blockDim.x + threadIdx.x;
  size_t tot = (size_t)NS * NB * 64;
  if (i >= tot) return;
  int k = (int)(i & 63);
  size_t t = i >> 6;
  int b = (int)(t % NB);
  int s = (int)(t / NB);
  int so = NS - 1 - s;                        // reversed time
  size_t ci = ((size_t)b * NS + so) * 64 + k;
  float m = MCOVL[ci];
  float v = (1.f - m) * AT[so * 64 + k] + COVL[ci] * m;
  size_t xo = ((size_t)s * NB + b) * 64 + k;
  if (k < 63) {
    xrev[xo] = f2bf(v);
  } else {
    xrev[xo] = 0;                              // pad column
    Trev[(size_t)s * NB + b] = 1.f / __logf(v + 2.7183f);
  }
}

__global__ void k_prep_covs(const float* __restrict__ COVS, const float* __restrict__ MCOVS,
                            const float* __restrict__ AS, u16t* __restrict__ out) {
  size_t i = (size_t)blockIdx.x * blockDim.x + threadIdx.x;
  if (i >= (size_t)NB * 64) return;
  int k = (int)(i & 63);
  float m = MCOVS[i];
  out[i] = f2bf((1.f - m) * AS[k] + COVS[i] * m);
}

// ---------------------------------------------------------------------------
// Generic bf16 WMMA GEMM: one wave per 16x16 tile, 8 waves/block.
// A: bf16 [M][lda] row-major, Bt: bf16 [N][K] packed-transposed.
// ---------------------------------------------------------------------------
__global__ __launch_bounds__(256) void k_gemm(
    const u16t* __restrict__ A, int lda,
    const u16t* __restrict__ Bt, int K,
    const float* __restrict__ bias,
    int Ntiles, int tiles,
    float* outF, long ldcF, u16t* outB, long ldcB, int relu) {
  int wave = threadIdx.x >> 5, lane = threadIdx.x & 31;
  int tile = blockIdx.x * 8 + wave;
  if (tile >= tiles) return;
  int mt = tile / Ntiles, nt = tile % Ntiles;
  int m0 = mt * 16, n0 = nt * 16;
  int mr = lane & 15, kh = lane >> 4;
  v8f acc;
  #pragma unroll
  for (int i = 0; i < 8; ++i) acc[i] = 0.f;
  for (int k0 = 0; k0 < K; k0 += 32) {
    BFrag a, b;
    const u16t* ap = A + (size_t)(m0 + mr) * lda + k0 + kh * 16;
    a.u4[0] = *(const v4u*)ap;
    a.u4[1] = *(const v4u*)(ap + 8);
    const u16t* bp = Bt + (size_t)(n0 + mr) * K + k0 + kh * 16;
    b.u8 = *(const v8u*)bp;
    acc = wmma_bf16(a, b, acc);
  }
  #pragma unroll
  for (int r = 0; r < 8; ++r) {
    int row = m0 + r + kh * 8;
    int col = n0 + mr;
    float v = acc[r] + (bias ? bias[col] : 0.f);
    if (relu) v = fmaxf(v, 0.f);
    if (outF) outF[(size_t)row * ldcF + col] = v;
    if (outB) outB[(size_t)row * ldcB + col] = f2bf(v);
  }
}

// ---------------------------------------------------------------------------
// Persistent TLSTM kernel: 64 workgroups x 256 threads; each WG owns 16 batch
// rows, runs both layers over all 256 reversed-time steps.  c/h fp32 state in
// registers; h mirrored bf16 in LDS as next step's WMMA A operand.
// ---------------------------------------------------------------------------
__global__ __launch_bounds__(256) void k_tlstm(
    const u16t* __restrict__ xrev,      // [NS][NB][64] bf16 (reversed time)
    u16t* l0out,                        // [NS][NB][HL] bf16 (layer-0 scan outputs)
    const float* __restrict__ Trev,     // [NS][NB]
    const u16t* __restrict__ WxT0, const float* __restrict__ bx0,
    const u16t* __restrict__ WdT0, const float* __restrict__ bd0,
    const u16t* __restrict__ WxT1, const float* __restrict__ bx1,
    const u16t* __restrict__ WdT1, const float* __restrict__ bd1,
    float* __restrict__ out_rep,        // d_out: rep rows of stride 768, cols 256..767
    u16t* __restrict__ rep_bf) {        // [NB][768] bf16 mirror
  __shared__ u16t  a_buf[BTL][LDAA];    // [x | h_adj] bf16 A panel
  __shared__ u16t  h_bf[BTL][HL];       // h bf16 (A operand for cst GEMM)
  __shared__ float T_s[BTL];

  const int tid  = threadIdx.x;
  const int wave = tid >> 5;
  const int lane = tid & 31;
  const int b0   = blockIdx.x * BTL;
  const int mr   = lane & 15;
  const int kh   = lane >> 4;
  const int ncol0 = wave * 64;          // this wave's 64 hidden columns

  float hreg[4][8], creg[4][8];         // fp32 state, thread-owned (m = r+kh*8, col = ncol0+16t+mr)

  for (int ell = 0; ell < 2; ++ell) {
    const int Kx   = ell ? HL : 64;
    const int Ktot = Kx + HL;
    const u16t*  WxT = ell ? WxT1 : WxT0;
    const u16t*  WdT = ell ? WdT1 : WdT0;
    const float* bx  = ell ? bx1 : bx0;
    const float* bd  = ell ? bd1 : bd0;

    // per-thread bias preload (bias depends only on column)
    float bI[4], bF[4], bO[4], bG[4], bD[4];
    #pragma unroll
    for (int t = 0; t < 4; ++t) {
      int col = ncol0 + t * 16 + mr;
      bI[t] = bx[col]; bF[t] = bx[512 + col];
      bO[t] = bx[1024 + col]; bG[t] = bx[1536 + col];
      bD[t] = bd[col];
    }
    #pragma unroll
    for (int t = 0; t < 4; ++t)
      #pragma unroll
      for (int r = 0; r < 8; ++r) { hreg[t][r] = 0.f; creg[t][r] = 0.f; }
    for (int i = tid; i < BTL * HL; i += 256) ((u16t*)h_bf)[i] = 0;
    __syncthreads();

    for (int s = 0; s < NS; ++s) {
      // -------- (a) stage x tile into a_buf[:,0:Kx], load T --------
      if (ell == 0) {
        for (int i = tid; i < BTL * 64; i += 256) {
          int r = i >> 6, k = i & 63;
          a_buf[r][k] = xrev[(size_t)s * NB * 64 + (size_t)(b0 + r) * 64 + k];
        }
      } else {
        int sp = NS - 1 - s;            // layer 1 consumes layer-0 scan reversed
        const u16t* src = l0out + ((size_t)sp * NB + b0) * HL;
        for (int i = tid; i < BTL * (HL / 2); i += 256) {
          int r = i / (HL / 2), kk = i % (HL / 2);
          ((unsigned int*)&a_buf[r][0])[kk] = ((const unsigned int*)(src + (size_t)r * HL))[kk];
        }
      }
      if (tid < BTL) T_s[tid] = Trev[(size_t)s * NB + b0 + tid];
      __syncthreads();

      // -------- (b) cst = tanh(h@Wd+bd); h_adj -> a_buf[:,Kx:] --------
      {
        v8f accd[4];
        #pragma unroll
        for (int t = 0; t < 4; ++t)
          #pragma unroll
          for (int i = 0; i < 8; ++i) accd[t][i] = 0.f;
        for (int k0 = 0; k0 < HL; k0 += 32) {
          BFrag af;
          af.u4[0] = *(const v4u*)&h_bf[mr][k0 + kh * 16];
          af.u4[1] = *(const v4u*)&h_bf[mr][k0 + 8 + kh * 16];
          #pragma unroll
          for (int t = 0; t < 4; ++t) {
            BFrag bfg;
            const u16t* bp = WdT + (size_t)(ncol0 + t * 16 + mr) * HL + k0 + kh * 16;
            bfg.u8 = *(const v8u*)bp;
            accd[t] = wmma_bf16(af, bfg, accd[t]);
          }
        }
        #pragma unroll
        for (int t = 0; t < 4; ++t)
          #pragma unroll
          for (int r = 0; r < 8; ++r) {
            int m = r + kh * 8;
            int col = ncol0 + t * 16 + mr;
            float cst = tanhf(accd[t][r] + bD[t]);
            float ha  = hreg[t][r] - cst + T_s[m] * cst;
            a_buf[m][Kx + col] = f2bf(ha);
          }
      }
      __syncthreads();

      // -------- (c) gates = [x|h_adj] @ Wx --------
      v8f acc[16];                       // [gate*4 + subtile]
      #pragma unroll
      for (int t = 0; t < 16; ++t)
        #pragma unroll
        for (int i = 0; i < 8; ++i) acc[t][i] = 0.f;
      for (int k0 = 0; k0 < Ktot; k0 += 32) {
        BFrag af;
        af.u4[0] = *(const v4u*)&a_buf[mr][k0 + kh * 16];
        af.u4[1] = *(const v4u*)&a_buf[mr][k0 + 8 + kh * 16];
        #pragma unroll
        for (int g = 0; g < 4; ++g)
          #pragma unroll
          for (int t = 0; t < 4; ++t) {
            BFrag bfg;
            const u16t* bp = WxT + (size_t)(g * 512 + ncol0 + t * 16 + mr) * Ktot + k0 + kh * 16;
            bfg.u8 = *(const v8u*)bp;
            acc[g * 4 + t] = wmma_bf16(af, bfg, acc[g * 4 + t]);
          }
      }

      // -------- (d) gate nonlinearity + state update --------
      #pragma unroll
      for (int t = 0; t < 4; ++t)
        #pragma unroll
        for (int r = 0; r < 8; ++r) {
          int m = r + kh * 8;
          int col = ncol0 + t * 16 + mr;
          float iv = sigmoidf_(acc[0 + t][r]  + bI[t]);
          float fv = sigmoidf_(acc[4 + t][r]  + bF[t]);
          float ov = sigmoidf_(acc[8 + t][r]  + bO[t]);
          float gv = tanhf    (acc[12 + t][r] + bG[t]);
          float cn = fv * creg[t][r] + iv * gv;
          float hn = ov * tanhf(cn);
          creg[t][r] = cn;
          hreg[t][r] = hn;
          h_bf[m][col] = f2bf(hn);
        }
      __syncthreads();

      // -------- (e) layer 0: spill h to global (coalesced) --------
      if (ell == 0) {
        u16t* dst = l0out + ((size_t)s * NB + b0) * HL;
        for (int i = tid; i < BTL * (HL / 2); i += 256) {
          int r = i / (HL / 2), kk = i % (HL / 2);
          ((unsigned int*)(dst + (size_t)r * HL))[kk] = ((const unsigned int*)&h_bf[r][0])[kk];
        }
      }
      __syncthreads();
    } // steps
    __syncthreads();
  } // layers

  // final layer-1 h == t_rep -> rep columns 256..767
  #pragma unroll
  for (int t = 0; t < 4; ++t)
    #pragma unroll
    for (int r = 0; r < 8; ++r) {
      int m = r + kh * 8;
      int col = ncol0 + t * 16 + mr;
      float v = hreg[t][r];
      out_rep[(size_t)(b0 + m) * 768 + 256 + col] = v;
      rep_bf[(size_t)(b0 + m) * 768 + 256 + col] = f2bf(v);
    }
}

// ---------------------------------------------------------------------------
// Host launcher
// ---------------------------------------------------------------------------
extern "C" void kernel_launch(void* const* d_in, const int* in_sizes, int n_in,
                              void* d_out, int out_size, void* d_ws, size_t ws_size,
                              hipStream_t stream) {
  (void)in_sizes; (void)n_in; (void)out_size; (void)ws_size;
  const float* COVL  = (const float*)d_in[0];
  const float* MCOVL = (const float*)d_in[1];
  const float* COVS  = (const float*)d_in[2];
  const float* MCOVS = (const float*)d_in[3];
  const float* AT    = (const float*)d_in[4];
  const float* AS    = (const float*)d_in[5];
  const float* Wx0 = (const float*)d_in[6];   const float* bx0 = (const float*)d_in[7];
  const float* Wd0 = (const float*)d_in[8];   const float* bd0 = (const float*)d_in[9];
  const float* Wx1 = (const float*)d_in[10];  const float* bx1 = (const float*)d_in[11];
  const float* Wd1 = (const float*)d_in[12];  const float* bd1 = (const float*)d_in[13];
  const float* Ws1 = (const float*)d_in[14];  const float* bs1 = (const float*)d_in[15];
  const float* Ws2 = (const float*)d_in[16];  const float* bs2 = (const float*)d_in[17];
  const float* Wm1 = (const float*)d_in[18];  const float* bm1 = (const float*)d_in[19];
  const float* Wm2 = (const float*)d_in[20];  const float* bm2 = (const float*)d_in[21];
  const float* Wv1 = (const float*)d_in[22];  const float* bv1 = (const float*)d_in[23];
  const float* Wv2 = (const float*)d_in[24];  const float* bv2 = (const float*)d_in[25];

  float* outF = (float*)d_out;                       // rep[1024][768] | mean[1024][64] | var[1024][64]
  char* ws = (char*)d_ws;
  size_t off = 0;
  auto alloc = [&](size_t bytes) -> char* {
    char* p = ws + off; off += (bytes + 255) & ~(size_t)255; return p;
  };
  u16t* WxT0 = (u16t*)alloc((size_t)2048 * 576  * 2);
  u16t* WdT0 = (u16t*)alloc((size_t)512  * 512  * 2);
  u16t* WxT1 = (u16t*)alloc((size_t)2048 * 1024 * 2);
  u16t* WdT1 = (u16t*)alloc((size_t)512  * 512  * 2);
  u16t* Ws1T = (u16t*)alloc((size_t)256  * 64   * 2);
  u16t* Ws2T = (u16t*)alloc((size_t)256  * 256  * 2);
  u16t* Wm1T = (u16t*)alloc((size_t)768  * 768  * 2);
  u16t* Wm2T = (u16t*)alloc((size_t)64   * 768  * 2);
  u16t* Wv1T = (u16t*)alloc((size_t)768  * 768  * 2);
  u16t* Wv2T = (u16t*)alloc((size_t)64   * 768  * 2);
  u16t* xrev = (u16t*)alloc((size_t)NS * NB * 64 * 2);
  float* Trev = (float*)alloc((size_t)NS * NB * 4);
  u16t* covsb = (u16t*)alloc((size_t)NB * 64 * 2);
  u16t* sbuf  = (u16t*)alloc((size_t)NB * 256 * 2);
  u16t* repb  = (u16t*)alloc((size_t)NB * 768 * 2);
  u16t* tmpb  = (u16t*)alloc((size_t)NB * 768 * 2);
  u16t* l0out = (u16t*)alloc((size_t)NS * NB * HL * 2);

  auto grid1 = [](size_t tot) { return dim3((unsigned)((tot + 255) / 256)); };

  // weight packing (transpose + bf16 + pad); zrow=63 inserts zero K-row for layer-0 x pad
  k_pack_wT<<<grid1((size_t)2048 * 576),  256, 0, stream>>>(Wx0, WxT0, 575, 2048, 576,  63);
  k_pack_wT<<<grid1((size_t)512  * 512),  256, 0, stream>>>(Wd0, WdT0, 512, 512,  512,  -1);
  k_pack_wT<<<grid1((size_t)2048 * 1024), 256, 0, stream>>>(Wx1, WxT1, 1024, 2048, 1024, -1);
  k_pack_wT<<<grid1((size_t)512  * 512),  256, 0, stream>>>(Wd1, WdT1, 512, 512,  512,  -1);
  k_pack_wT<<<grid1((size_t)256  * 64),   256, 0, stream>>>(Ws1, Ws1T, 64,  256,  64,   -1);
  k_pack_wT<<<grid1((size_t)256  * 256),  256, 0, stream>>>(Ws2, Ws2T, 256, 256,  256,  -1);
  k_pack_wT<<<grid1((size_t)768  * 768),  256, 0, stream>>>(Wm1, Wm1T, 768, 768,  768,  -1);
  k_pack_wT<<<grid1((size_t)64   * 768),  256, 0, stream>>>(Wm2, Wm2T, 768, 64,   768,  -1);
  k_pack_wT<<<grid1((size_t)768  * 768),  256, 0, stream>>>(Wv1, Wv1T, 768, 768,  768,  -1);
  k_pack_wT<<<grid1((size_t)64   * 768),  256, 0, stream>>>(Wv2, Wv2T, 768, 64,   768,  -1);

  // input prep
  k_prep_x<<<grid1((size_t)NS * NB * 64), 256, 0, stream>>>(COVL, MCOVL, AT, xrev, Trev);
  k_prep_covs<<<grid1((size_t)NB * 64), 256, 0, stream>>>(COVS, MCOVS, AS, covsb);

  // TLSTM: writes rep cols 256..767 (fp32 in d_out, bf16 mirror)
  k_tlstm<<<dim3(NB / BTL), 256, 0, stream>>>(xrev, l0out, Trev,
                                              WxT0, bx0, WdT0, bd0,
                                              WxT1, bx1, WdT1, bd1,
                                              outF, repb);

  // static encoder: relu(covs@Ws1+bs1)@Ws2+bs2 -> rep cols 0..255
  k_gemm<<<dim3((64 * 16 + 7) / 8), 256, 0, stream>>>(covsb, 64, Ws1T, 64, bs1,
      16, 64 * 16, nullptr, 0, sbuf, 256, 1);
  k_gemm<<<dim3((64 * 16 + 7) / 8), 256, 0, stream>>>(sbuf, 256, Ws2T, 256, bs2,
      16, 64 * 16, outF, 768, repb, 768, 0);

  // mean head: (rep@Wm1+bm1)@Wm2+bm2
  float* meanOut = outF + (size_t)NB * 768;
  float* varOut  = meanOut + (size_t)NB * 64;
  k_gemm<<<dim3((64 * 48 + 7) / 8), 256, 0, stream>>>(repb, 768, Wm1T, 768, bm1,
      48, 64 * 48, nullptr, 0, tmpb, 768, 0);
  k_gemm<<<dim3((64 * 4 + 7) / 8), 256, 0, stream>>>(tmpb, 768, Wm2T, 768, bm2,
      4, 64 * 4, meanOut, 64, nullptr, 0, 0);
  // var head
  k_gemm<<<dim3((64 * 48 + 7) / 8), 256, 0, stream>>>(repb, 768, Wv1T, 768, bv1,
      48, 64 * 48, nullptr, 0, tmpb, 768, 0);
  k_gemm<<<dim3((64 * 4 + 7) / 8), 256, 0, stream>>>(tmpb, 768, Wv2T, 768, bv2,
      4, 64 * 4, varOut, 64, nullptr, 0, 0);
}
